// VTN_21758304321907
// MI455X (gfx1250) — compile-verified, matrix operand order checked by reading, not run on hardware
//
#include <hip/hip_runtime.h>

// ---------------------------------------------------------------------------
// MI455X (gfx1250, wave32) transformer-VAE forward pass.
// Dead code eliminated (LSTM, prior heads, KL — multiplied by 0 / unused).
// All matmuls on v_wmma_f32_16x16x32_bf16; fp32 residual stream; fp32->bf16
// conversion fused into vectorized LDS staging; GEMM is register-pipelined.
// Staging uses uniform-base + 32-bit offsets so loads select the
// global_load saddr form (no FLAT fallback, no DScnt pollution).
// ---------------------------------------------------------------------------

typedef __attribute__((ext_vector_type(16))) __bf16 v16bf;
typedef __attribute__((ext_vector_type(4)))  __bf16 v4bf;
typedef __attribute__((ext_vector_type(2)))  __bf16 v2bf;
typedef __attribute__((ext_vector_type(8)))  float  v8f;

#define BT_  4096      // B*T
#define C_   512
#define T_   1024
#define H_   8
#define D_   64
#define V_   32000

__device__ __forceinline__ v8f wmma_bf16(v16bf a, v16bf b, v8f c) {
  return __builtin_amdgcn_wmma_f32_16x16x32_bf16(false, a, false, b,
                                                 (short)0, c, false, false);
}

// Load one 16x32 bf16 fragment (A-layout; identical layout serves as the
// n-major B fragment) from an LDS tile stored row-major with leading dim ldk.
// Lane l: row = l&15, half = l>>4.  VGPR v holds packed K pair:
//   v<4 : K = half*8 + 2v,2v+1      v>=4 : K = 16 + half*8 + 2(v-4), +1
__device__ __forceinline__ v16bf load_frag(const __bf16* tile, int row0,
                                           int koff, int ldk) {
  const int lane = threadIdx.x & 31;
  const __bf16* p = tile + (row0 + (lane & 15)) * ldk + koff + ((lane >> 4) << 3);
  union { v16bf v; unsigned u[8]; } f;
  const unsigned* lo = (const unsigned*)p;
  const unsigned* hi = (const unsigned*)(p + 16);
  f.u[0] = lo[0]; f.u[1] = lo[1]; f.u[2] = lo[2]; f.u[3] = lo[3];
  f.u[4] = hi[0]; f.u[5] = hi[1]; f.u[6] = hi[2]; f.u[7] = hi[3];
  return f.v;
}

// Native <2 x bf16> store: one v_cvt_pk_bf16_f32 + one ds_store_b32.
__device__ __forceinline__ void store_bf16x2(__bf16* dst, float lo, float hi) {
  v2bf p;
  p.x = (__bf16)lo; p.y = (__bf16)hi;
  *(v2bf*)dst = p;
}

// Native <4 x bf16> store: two packed converts + one ds_store_b64.
__device__ __forceinline__ void store_bf16x4(__bf16* dst, float4 t) {
  v4bf p;
  p.x = (__bf16)t.x; p.y = (__bf16)t.y;
  p.z = (__bf16)t.z; p.w = (__bf16)t.w;
  *(v4bf*)dst = p;
}

__device__ __forceinline__ float gelu_exact(float x) {
  return 0.5f * x * (1.0f + erff(x * 0.70710678118654752f));
}

// ---------------------------------------------------------------------------
// GEMM: C[M,N] = act(A[M,K] @ B[K,N] + bias) + resid     (fp32 in/out,
// bf16 WMMA core).  Tiles: 128x128x32, 256 threads = 8 waves, wave tile 32x64.
// Register-pipelined staging: next K-tile is loaded to registers while the
// current tile's WMMAs run out of LDS.
// ---------------------------------------------------------------------------
__global__ __launch_bounds__(256) void gemm_kernel(
    const float* __restrict__ A, const float* __restrict__ B,
    const float* __restrict__ bias, const float* __restrict__ resid,
    float* __restrict__ C, int M, int N, int K, int act) {
  __shared__ __bf16 As[128][32];   // m-major, k contiguous
  __shared__ __bf16 Bs[128][32];   // n-major, k contiguous (B transposed)

  const int tid  = threadIdx.x;
  const int m0   = blockIdx.y * 128;
  const int n0   = blockIdx.x * 128;
  const int wave = tid >> 5;
  const int wm   = (wave & 3) * 32;   // wave row offset in tile
  const int wn   = (wave >> 2) * 64;  // wave col offset in tile

  // Per-thread staging map: uniform base + 32-bit element offsets
  // (global_load saddr form; all byte offsets < 2^31).
  //   A: vec v = tid + i*256 (i<4): r = v>>3, c4 = (v&7)*4      (coalesced: k)
  //   B: v = tid + i*256 (i<2): kp = v>>5 (k-row pair), nn4 = (v&31)*4
  int aoff[4];
  __bf16* asd[4];
#pragma unroll
  for (int i = 0; i < 4; ++i) {
    int vi = tid + i * 256;
    int r  = vi >> 3, c4 = (vi & 7) << 2;
    aoff[i] = (m0 + r) * K + c4;
    asd[i]  = &As[r][c4];
  }
  int boff[2];
  __bf16* bsd[2];
#pragma unroll
  for (int i = 0; i < 2; ++i) {
    int vi  = tid + i * 256;          // 0..511
    int kp  = vi >> 5;                // 0..15  (k rows 2kp, 2kp+1)
    int nn4 = (vi & 31) << 2;         // 0..124
    boff[i] = (2 * kp) * N + n0 + nn4;
    bsd[i]  = &Bs[nn4][2 * kp];
  }
  const int bstep = 32 * N;

  float4 ar[4], br0[2], br1[2];
#pragma unroll
  for (int i = 0; i < 4; ++i) ar[i] = *(const float4*)(A + aoff[i]);
#pragma unroll
  for (int i = 0; i < 2; ++i) {
    br0[i] = *(const float4*)(B + boff[i]);
    br1[i] = *(const float4*)(B + boff[i] + N);
  }

  v8f acc[2][4] = {};

  for (int k0 = 0; k0 < K; k0 += 32) {
    // Commit staged registers to LDS (fp32 -> packed bf16).
#pragma unroll
    for (int i = 0; i < 4; ++i) store_bf16x4(asd[i], ar[i]);
#pragma unroll
    for (int i = 0; i < 2; ++i) {
      store_bf16x2(bsd[i] + 0 * 32, br0[i].x, br1[i].x);
      store_bf16x2(bsd[i] + 1 * 32, br0[i].y, br1[i].y);
      store_bf16x2(bsd[i] + 2 * 32, br0[i].z, br1[i].z);
      store_bf16x2(bsd[i] + 3 * 32, br0[i].w, br1[i].w);
    }
    __syncthreads();

    // Prefetch next K-tile into registers; latency hides behind the WMMAs.
    if (k0 + 32 < K) {
#pragma unroll
      for (int i = 0; i < 4; ++i) {
        aoff[i] += 32;
        ar[i] = *(const float4*)(A + aoff[i]);
      }
#pragma unroll
      for (int i = 0; i < 2; ++i) {
        boff[i] += bstep;
        br0[i] = *(const float4*)(B + boff[i]);
        br1[i] = *(const float4*)(B + boff[i] + N);
      }
      if (k0 + 64 < K)  // warm L2 one tile further ahead
        __builtin_prefetch(B + boff[0] + bstep, 0, 1);
    }

    v16bf af[2], bf[4];
#pragma unroll
    for (int mi = 0; mi < 2; ++mi) af[mi] = load_frag(&As[0][0], wm + mi * 16, 0, 32);
#pragma unroll
    for (int ni = 0; ni < 4; ++ni) bf[ni] = load_frag(&Bs[0][0], wn + ni * 16, 0, 32);
#pragma unroll
    for (int mi = 0; mi < 2; ++mi)
#pragma unroll
      for (int ni = 0; ni < 4; ++ni)
        acc[mi][ni] = wmma_bf16(af[mi], bf[ni], acc[mi][ni]);
    __syncthreads();
  }

  const int lane = tid & 31, half = lane >> 4, cn = lane & 15;
#pragma unroll
  for (int mi = 0; mi < 2; ++mi)
#pragma unroll
    for (int ni = 0; ni < 4; ++ni)
#pragma unroll
      for (int r = 0; r < 8; ++r) {
        int row = m0 + wm + mi * 16 + r + half * 8;
        int col = n0 + wn + ni * 16 + cn;
        float v = acc[mi][ni][r];
        if (bias)  v += bias[col];
        if (act)   v = gelu_exact(v);
        if (resid) v += resid[(size_t)row * N + col];
        C[(size_t)row * N + col] = v;
      }
}

// ---------------------------------------------------------------------------
// Causal flash attention.  One workgroup = (b, h, 64-row q block), 128 threads
// = 4 waves; each wave owns a 16-row q strip.  S = Q·K^T and O += P·V on WMMA.
// q/k/v layout: [B*T, C] with head h occupying columns h*64..h*64+63.
// ---------------------------------------------------------------------------
__global__ __launch_bounds__(128) void attn_kernel(
    const float* __restrict__ q, const float* __restrict__ k,
    const float* __restrict__ v, float* __restrict__ y) {
  __shared__ __bf16 Qs[64][64];        // m-major, d contiguous   (A for S)
  __shared__ __bf16 Ks[64][64];        // kv-major, d contiguous  (B for S, n-major)
  __shared__ __bf16 Vts[64][64];       // d-major, kv contiguous  (B for O, n-major)
  __shared__ __bf16 Ps[4][16][64];     // per-wave P strip        (A for O)

  const int tid = threadIdx.x, lane = tid & 31, wave = tid >> 5;
  const int half = lane >> 4, cn = lane & 15;
  const int qb = blockIdx.x, h = blockIdx.y, b = blockIdx.z;
  const int q0 = qb * 64;

  // Stage Q (vectorized: float4 loads, b64 LDS stores).
#pragma unroll
  for (int i = 0; i < 8; ++i) {
    int e = tid + i * 128;
    int rr = e >> 4, dd4 = (e & 15) << 2;
    float4 t = *(const float4*)&q[(b * T_ + q0 + rr) * C_ + h * 64 + dd4];
    store_bf16x4(&Qs[rr][dd4], t);
  }
  __syncthreads();

  v16bf aq0 = load_frag(&Qs[0][0], wave * 16, 0, 64);
  v16bf aq1 = load_frag(&Qs[0][0], wave * 16, 32, 64);

  v8f o[4] = {};
  float mrow[8], lrow[8];
#pragma unroll
  for (int r = 0; r < 8; ++r) { mrow[r] = -1e30f; lrow[r] = 0.f; }

  for (int j = 0; j <= qb; ++j) {
    // K tile: natural layout, b64 stores.
#pragma unroll
    for (int i = 0; i < 8; ++i) {
      int e = tid + i * 128;
      int kv = e >> 4, dd4 = (e & 15) << 2;
      float4 tk = *(const float4*)&k[(b * T_ + j * 64 + kv) * C_ + h * 64 + dd4];
      store_bf16x4(&Ks[kv][dd4], tk);
    }
    // V tile transposed: kv-row pairs -> packed <2 x bf16> b32 stores.
#pragma unroll
    for (int i = 0; i < 4; ++i) {
      int e = tid + i * 128;            // 0..511
      int kvp = e >> 4;                 // 0..31 (kv rows 2kvp, 2kvp+1)
      int dd4 = (e & 15) << 2;
      int g = (b * T_ + j * 64 + 2 * kvp) * C_ + h * 64 + dd4;
      float4 t0 = *(const float4*)&v[g];
      float4 t1 = *(const float4*)&v[g + C_];
      __bf16* d = &Vts[dd4][2 * kvp];
      store_bf16x2(d + 0 * 64, t0.x, t1.x);
      store_bf16x2(d + 1 * 64, t0.y, t1.y);
      store_bf16x2(d + 2 * 64, t0.z, t1.z);
      store_bf16x2(d + 3 * 64, t0.w, t1.w);
    }
    __syncthreads();

    v8f s[4] = {};
#pragma unroll
    for (int ni = 0; ni < 4; ++ni) {
      s[ni] = wmma_bf16(aq0, load_frag(&Ks[0][0], ni * 16, 0, 64), s[ni]);
      s[ni] = wmma_bf16(aq1, load_frag(&Ks[0][0], ni * 16, 32, 64), s[ni]);
    }
    // scale + causal mask (only the diagonal block needs masking)
#pragma unroll
    for (int ni = 0; ni < 4; ++ni)
#pragma unroll
      for (int r = 0; r < 8; ++r) {
        float sv = s[ni][r] * 0.125f;               // 1/sqrt(64)
        if (j == qb) {
          int scol = j * 64 + ni * 16 + cn;
          int trow = q0 + wave * 16 + r + half * 8;
          if (scol > trow) sv = -1e30f;
        }
        s[ni][r] = sv;
      }
    // online softmax, row-wise (row = r + 8*half; 16-lane half-group shares row)
#pragma unroll
    for (int r = 0; r < 8; ++r) {
      float rm = fmaxf(fmaxf(s[0][r], s[1][r]), fmaxf(s[2][r], s[3][r]));
#pragma unroll
      for (int msk = 8; msk >= 1; msk >>= 1) rm = fmaxf(rm, __shfl_xor(rm, msk, 32));
      float mn = fmaxf(mrow[r], rm);
      float alpha = expf(mrow[r] - mn);
      mrow[r] = mn;
      float rs = 0.f;
#pragma unroll
      for (int ni = 0; ni < 4; ++ni) {
        float p = expf(s[ni][r] - mn);
        rs += p;
        Ps[wave][r + half * 8][ni * 16 + cn] = (__bf16)p;
      }
#pragma unroll
      for (int msk = 8; msk >= 1; msk >>= 1) rs += __shfl_xor(rs, msk, 32);
      lrow[r] = lrow[r] * alpha + rs;
#pragma unroll
      for (int ni = 0; ni < 4; ++ni) o[ni][r] *= alpha;
    }
    // O += P·V  (per-wave LDS RAW: DS is in-order within a wave)
    v16bf ap0 = load_frag(&Ps[wave][0][0], 0, 0, 64);
    v16bf ap1 = load_frag(&Ps[wave][0][0], 0, 32, 64);
#pragma unroll
    for (int ni = 0; ni < 4; ++ni) {
      o[ni] = wmma_bf16(ap0, load_frag(&Vts[0][0], ni * 16, 0, 64), o[ni]);
      o[ni] = wmma_bf16(ap1, load_frag(&Vts[0][0], ni * 16, 32, 64), o[ni]);
    }
    __syncthreads();
  }

#pragma unroll
  for (int ni = 0; ni < 4; ++ni)
#pragma unroll
    for (int r = 0; r < 8; ++r) {
      float ov = o[ni][r] / lrow[r];
      y[(b * T_ + q0 + wave * 16 + r + half * 8) * C_ + h * 64 + ni * 16 + cn] = ov;
    }
}

// ---------------------------------------------------------------------------
// LayerNorm over C=512: one row per wave (8 rows / 256-thread block), float4.
// ---------------------------------------------------------------------------
__global__ __launch_bounds__(256) void ln_kernel(
    const float* __restrict__ x, const float* __restrict__ w,
    const float* __restrict__ b, float* __restrict__ out) {
  int row  = blockIdx.x * 8 + (threadIdx.x >> 5);
  int lane = threadIdx.x & 31;
  const float4* xr = (const float4*)(x + (size_t)row * C_);
  const float4* w4 = (const float4*)w;
  const float4* b4 = (const float4*)b;
  float4* o4 = (float4*)(out + (size_t)row * C_);
  float s = 0.f, sq = 0.f;
  float4 t[4];
#pragma unroll
  for (int i = 0; i < 4; ++i) {
    t[i] = xr[lane + i * 32];
    s  += t[i].x + t[i].y + t[i].z + t[i].w;
    sq += t[i].x * t[i].x + t[i].y * t[i].y + t[i].z * t[i].z + t[i].w * t[i].w;
  }
#pragma unroll
  for (int m = 16; m >= 1; m >>= 1) { s += __shfl_xor(s, m, 32); sq += __shfl_xor(sq, m, 32); }
  float mean = s * (1.f / C_);
  float var  = sq * (1.f / C_) - mean * mean;
  float rstd = rsqrtf(var + 1e-5f);
#pragma unroll
  for (int i = 0; i < 4; ++i) {
    float4 ww = w4[lane + i * 32], bb = b4[lane + i * 32], r;
    r.x = (t[i].x - mean) * rstd * ww.x + bb.x;
    r.y = (t[i].y - mean) * rstd * ww.y + bb.y;
    r.z = (t[i].z - mean) * rstd * ww.z + bb.z;
    r.w = (t[i].w - mean) * rstd * ww.w + bb.w;
    o4[lane + i * 32] = r;
  }
}

__global__ __launch_bounds__(128) void embed_kernel(const int* __restrict__ idx,
                                                    const float* __restrict__ emb,
                                                    float* __restrict__ x) {
  int row = blockIdx.x;
  int t = idx[row];
  const float4* e4 = (const float4*)(emb + (size_t)t * C_);
  float4* x4 = (float4*)(x + (size_t)row * C_);
  x4[threadIdx.x] = e4[threadIdx.x];   // 128 threads * float4 = 512 floats
}

__global__ void z_kernel(const float4* __restrict__ mu, const float4* __restrict__ ls,
                         const float4* __restrict__ noise, float4* __restrict__ z, int n4) {
  int i = blockIdx.x * 256 + threadIdx.x;
  if (i < n4) {
    float4 m = mu[i], l = ls[i], nz = noise[i], r;
    r.x = m.x + nz.x * expf(l.x);
    r.y = m.y + nz.y * expf(l.y);
    r.z = m.z + nz.z * expf(l.z);
    r.w = m.w + nz.w * expf(l.w);
    z[i] = r;
  }
}

__global__ void zero_loss(float* loss) {
  if (threadIdx.x == 0 && blockIdx.x == 0) *loss = 0.f;
}

// Row-wise CE: online logsumexp over V=32000, block reduction, atomic mean.
__global__ __launch_bounds__(256) void ce_kernel(
    const float* __restrict__ logits, const int* __restrict__ tgt,
    float* __restrict__ loss) {
  __shared__ float sm[256], ss[256];
  int row = blockIdx.x;
  const float* lr = logits + (size_t)row * V_;
  float m = -1e30f, s = 0.f;
  for (int c = threadIdx.x; c < V_; c += 256) {
    float x = lr[c];
    if (x > m) { s = s * expf(m - x) + 1.f; m = x; }
    else       { s += expf(x - m); }
  }
  sm[threadIdx.x] = m; ss[threadIdx.x] = s;
  __syncthreads();
  for (int off = 128; off > 0; off >>= 1) {
    if (threadIdx.x < off) {
      float m2 = sm[threadIdx.x + off], s2 = ss[threadIdx.x + off];
      float M = fmaxf(sm[threadIdx.x], m2);
      ss[threadIdx.x] = ss[threadIdx.x] * expf(sm[threadIdx.x] - M) + s2 * expf(m2 - M);
      sm[threadIdx.x] = M;
    }
    __syncthreads();
  }
  if (threadIdx.x == 0) {
    float lse = sm[0] + logf(ss[0]);
    float ce  = lse - lr[tgt[row]];
    atomicAdd(loss, ce * (1.0f / (float)BT_));
  }
}

// ---------------------------------------------------------------------------
// Orchestration.  d_in order (setup_inputs flattened):
//  0 idx  1 targets  2 noise  3 tok_emb
//  4..19  enc {ln1_w,ln1_b,Wq,bq,Wk,bk,Wv,bv,Wp,bp,ln2_w,ln2_b,W1,b1,W2,b2}
// 20..35  dec (same order)
// 36 mu_w 37 mu_b 38 sig_w 39 sig_b
// 40..43  lstm (dead)   44..47 prior heads (dead)
// 48 ln_f_w 49 ln_f_b 50 head_w
// ---------------------------------------------------------------------------
extern "C" void kernel_launch(void* const* d_in, const int* in_sizes, int n_in,
                              void* d_out, int out_size, void* d_ws, size_t ws_size,
                              hipStream_t stream) {
  const int*   idx     = (const int*)d_in[0];
  const int*   targets = (const int*)d_in[1];
  const float* noise   = (const float*)d_in[2];
  const float* tok_emb = (const float*)d_in[3];
  auto F = [&](int i) { return (const float*)d_in[i]; };

  // Workspace layout (fp32), ~101 MB total.
  float* xb  = (float*)d_ws;                 // [BT,C] residual stream
  float* hb  = xb  + (size_t)BT_ * C_;       // [BT,C] LN output
  float* qq  = hb  + (size_t)BT_ * C_;       // [BT,C]
  float* kk  = qq  + (size_t)BT_ * C_;       // [BT,C]
  float* vv  = kk  + (size_t)BT_ * C_;       // [BT,C]
  float* yb  = vv  + (size_t)BT_ * C_;       // [BT,C] attention out
  float* mid = yb  + (size_t)BT_ * C_;       // [BT,4C] MLP hidden
  float* mub = mid + (size_t)BT_ * 4 * C_;   // [BT,R]
  float* lsb = mub + (size_t)BT_ * C_;       // [BT,R]

  auto gemm = [&](const float* A, const float* B, const float* bias,
                  const float* res, float* Cout, int M, int N, int K, int act) {
    dim3 g(N / 128, M / 128);
    gemm_kernel<<<g, 256, 0, stream>>>(A, B, bias, res, Cout, M, N, K, act);
  };

  embed_kernel<<<BT_, 128, 0, stream>>>(idx, tok_emb, xb);

  auto run_stack = [&](int base) {
    for (int l = 0; l < 6; ++l) {
      size_t oC  = (size_t)l * C_;
      size_t oCC = (size_t)l * C_ * C_;
      size_t oC4 = (size_t)l * 4 * C_;
      size_t oW1 = (size_t)l * C_ * 4 * C_;
      ln_kernel<<<BT_ / 8, 256, 0, stream>>>(xb, F(base + 0) + oC, F(base + 1) + oC, hb);
      gemm(hb, F(base + 2) + oCC, F(base + 3) + oC, nullptr, qq, BT_, C_, C_, 0);
      gemm(hb, F(base + 4) + oCC, F(base + 5) + oC, nullptr, kk, BT_, C_, C_, 0);
      gemm(hb, F(base + 6) + oCC, F(base + 7) + oC, nullptr, vv, BT_, C_, C_, 0);
      attn_kernel<<<dim3(T_ / 64, H_, 4), 128, 0, stream>>>(qq, kk, vv, yb);
      gemm(yb, F(base + 8) + oCC, F(base + 9) + oC, xb, xb, BT_, C_, C_, 0);
      ln_kernel<<<BT_ / 8, 256, 0, stream>>>(xb, F(base + 10) + oC, F(base + 11) + oC, hb);
      gemm(hb, F(base + 12) + oW1, F(base + 13) + oC4, nullptr, mid, BT_, 4 * C_, C_, 1);
      gemm(mid, F(base + 14) + oW1, F(base + 15) + oC, xb, xb, BT_, C_, 4 * C_, 0);
    }
  };

  run_stack(4);                                          // encoder
  gemm(xb, F(36), F(37), nullptr, mub, BT_, C_, C_, 0);  // mu
  gemm(xb, F(38), F(39), nullptr, lsb, BT_, C_, C_, 0);  // log_sigma
  z_kernel<<<(BT_ * C_ / 4 + 255) / 256, 256, 0, stream>>>(
      (const float4*)mub, (const float4*)lsb, (const float4*)noise,
      (float4*)xb, BT_ * C_ / 4);
  run_stack(20);                                         // decoder
  ln_kernel<<<BT_ / 8, 256, 0, stream>>>(xb, F(48), F(49), hb);

  float* logits = (float*)d_out;
  gemm(hb, F(50), nullptr, nullptr, logits, BT_, V_, C_, 0);  // head
  float* loss = logits + (size_t)BT_ * V_;
  zero_loss<<<1, 32, 0, stream>>>(loss);
  ce_kernel<<<BT_, 256, 0, stream>>>(logits, targets, loss);
}
